// PodCritic_22883585753323
// MI455X (gfx1250) — matrix operand order repaired and checked by reading, not compiled, not run on hardware
//
#include <hip/hip_runtime.h>
#include <hip/hip_bf16.h>
#include <math.h>

typedef __attribute__((ext_vector_type(16))) _Float16 v16h;
typedef __attribute__((ext_vector_type(8)))  float    v8f;

#define B_TOT 4096
#define S_LEN 128
#define HID   64

// Packed f16 weight offsets (units: _Float16 elements)
// W1p : Kpad=64,  N=256 -> 16384
// W2p : K=256,    N=128 -> 32768
// Wihp: K=128,    N=256 -> 32768
// Whhp: K=64,     N=256 -> 16384   (total 98304 halves = 192 KB)
#define OFF_W1P   0
#define OFF_W2P   16384
#define OFF_WIHP  49152
#define OFF_WHHP  81920
#define WPK_HALVES 98304
#define WPK_BYTES  (WPK_HALVES * 2)   // 196608 B of dynamic LDS

// ---------------------------------------------------------------------------
// Pack a row-major f32 weight [Krows x N] into f16 WMMA-B fragment order.
// Tile (nt, kt) of 32x16; within a tile, element index = lane*16 + j where
//   col = nt*16 + (lane & 15)
//   k   = kt*32 + ((lane>>4)*16) + j        (j = 0..15, 16 consecutive K vals)
// Layout is [nt][kt][512] so the main kernel's inner-K loads are contiguous.
// ---------------------------------------------------------------------------
__global__ void pack_b_f16(const float* __restrict__ W, _Float16* __restrict__ dst,
                           int Krows, int Kpad, int N) {
  int total = Kpad * N;
  int tid = blockIdx.x * blockDim.x + threadIdx.x;
  if (tid >= total) return;
  int KT   = Kpad >> 5;               // k-tiles of 32
  int nt   = tid / (KT * 512);
  int rem  = tid - nt * (KT * 512);
  int kt   = rem >> 9;
  int l    = rem & 511;
  int lane = l >> 4;
  int j    = l & 15;
  int n = nt * 16 + (lane & 15);
  int k = kt * 32 + ((lane >> 4) << 4) + j;
  float v = (k < Krows) ? W[(size_t)k * N + n] : 0.0f;
  dst[tid] = (_Float16)v;
}

// A-fragment (16x32 f16) from an LDS activation tile act[16][ldk] at k=kbase.
// Contiguous K order (this code's choice): lane m = lane&15; half = lane>>4; lane holds K run
//   [kbase + half*16, +16) (matches packed B order).
__device__ __forceinline__ v16h load_A(const _Float16* act, int ldk, int kbase, int lane) {
  int m  = lane & 15;
  int h8 = (lane >> 4) << 4;
  const _Float16* row = act + m * ldk + kbase;
  v16h a;
#pragma unroll
  for (int j = 0; j < 8; ++j) a[j] = row[h8 + j];
#pragma unroll
  for (int j = 0; j < 8; ++j) a[8 + j] = row[8 + h8 + j];
  return a;
}

// B-fragment: 32 B per lane, contiguous, from LDS-resident packed weights.
__device__ __forceinline__ v16h load_B(const _Float16* wlds, int tile, int lane) {
  return *(const v16h*)(wlds + (size_t)tile * 512 + lane * 16);
}

__device__ __forceinline__ float sigmoidf_(float x) {
  return 1.0f / (1.0f + __expf(-x));
}

// ---------------------------------------------------------------------------
// Fused encoder + LSTM + value head.
// grid = B/16 blocks, 512 threads (16 wave32 waves). Block owns batch rows
// [b0, b0+16). Each wave owns one 16-column N-tile of every [16,256] GEMM.
// All packed weights (192 KB) live in CDNA5's 320 KB LDS for the whole block.
// ---------------------------------------------------------------------------
__global__ __launch_bounds__(512)
void pod_critic_fused(const float* __restrict__ self_obs,
                      const float* __restrict__ tm_obs,
                      const float* __restrict__ en_obs,
                      const float* __restrict__ cp_obs,
                      const float* __restrict__ b1,
                      const float* __restrict__ b2,
                      const float* __restrict__ bih,
                      const float* __restrict__ bhh,
                      const float* __restrict__ Wv,
                      const float* __restrict__ bv,
                      const _Float16* __restrict__ wpk,
                      float* __restrict__ out) {
  __shared__ _Float16 sx [16][64];    // padded concat input (f16)
  __shared__ _Float16 sh1[16][256];   // relu(x@W1+b1)
  __shared__ _Float16 semb[16][128];  // relu(h1@W2+b2)
  __shared__ float    sg [16][256];   // gates (f32)
  __shared__ _Float16 shh[16][64];    // h state (f16, feeds recurrent WMMA)
  __shared__ float    sc [16][64];    // c state (f32)
  __shared__ float    sb1[256], sb2[128], sbg[256], sWv[64];
  __shared__ float    sbv;
  extern __shared__ _Float16 sW[];    // 98304 halves: all packed weights

  const int tid  = threadIdx.x;
  const int lane = tid & 31;
  const int w    = tid >> 5;          // wave id 0..15
  const int b0   = blockIdx.x * 16;

  // ---- stage all packed weights into LDS once (16 B chunks, coalesced) ----
  {
    const int4* src = (const int4*)wpk;
    int4* dst = (int4*)sW;
    for (int i = tid; i < WPK_BYTES / 16; i += 512) dst[i] = src[i];
  }
  for (int i = tid; i < 256; i += 512) { sb1[i] = b1[i]; sbg[i] = bih[i] + bhh[i]; }
  for (int i = tid; i < 128; i += 512) sb2[i] = b2[i];
  if (tid < 64)  sWv[tid] = Wv[tid];
  if (tid == 0)  sbv = bv[0];
  for (int i = tid; i < 16 * 64; i += 512) {
    (&sc[0][0])[i]  = 0.0f;
    (&shh[0][0])[i] = (_Float16)0.0f;
  }
  __syncthreads();

  const _Float16* W1p = sW + OFF_W1P;
  const _Float16* W2p = sW + OFF_W2P;
  const _Float16* Wip = sW + OFF_WIHP;
  const _Float16* Whp = sW + OFF_WHHP;

  const int row0 = (lane >> 4) * 8;   // C-fragment: VGPR r -> M = row0 + r
  const int ncol = lane & 15;         //             N = wave_ntile*16 + ncol

  for (int t = 0; t < S_LEN; ++t) {
    // ---- stage x[16][64]: concat(self, tm, enemy-mean, cp) + zero pad -----
    for (int s = tid; s < 16 * 64; s += 512) {
      int r = s >> 6, c = s & 63;
      size_t base = (size_t)(b0 + r) * S_LEN + t;
      float v;
      if      (c < 15) v = self_obs[base * 15 + c];
      else if (c < 28) v = tm_obs[base * 13 + (c - 15)];
      else if (c < 41) { int cc = c - 28;
                         v = 0.5f * (en_obs[(base * 2 + 0) * 13 + cc] +
                                     en_obs[(base * 2 + 1) * 13 + cc]); }
      else if (c < 47) v = cp_obs[base * 6 + (c - 41)];
      else             v = 0.0f;
      (&sx[0][0])[s] = (_Float16)v;
    }
    // prefetch next timestep's inputs into cache (global_prefetch path)
    if (t + 1 < S_LEN && tid < 16) {
      size_t nb = (size_t)(b0 + tid) * S_LEN + (t + 1);
      __builtin_prefetch(&self_obs[nb * 15], 0, 1);
      __builtin_prefetch(&en_obs[nb * 2 * 13], 0, 1);
    }
    __syncthreads();

    // ---- GEMM1: sh1 = relu(sx @ W1p + b1)   K=64, N=256 -------------------
    {
      v8f acc = {0.f,0.f,0.f,0.f,0.f,0.f,0.f,0.f};
#pragma unroll
      for (int kt = 0; kt < 2; ++kt) {
        v16h a  = load_A(&sx[0][0], 64, kt * 32, lane);
        v16h bm = load_B(W1p, w * 2 + kt, lane);
        acc = __builtin_amdgcn_wmma_f32_16x16x32_f16(false, a, false, bm,
                                                     (short)0, acc, false, false);
      }
      int col = w * 16 + ncol;
#pragma unroll
      for (int r = 0; r < 8; ++r) {
        float v = acc[r] + sb1[col];
        sh1[row0 + r][col] = (_Float16)(v > 0.0f ? v : 0.0f);
      }
    }
    __syncthreads();

    // ---- GEMM2: semb = relu(sh1 @ W2p + b2)   K=256, N=128 (waves 0..7) ---
    if (w < 8) {
      v8f acc = {0.f,0.f,0.f,0.f,0.f,0.f,0.f,0.f};
#pragma unroll
      for (int kt = 0; kt < 8; ++kt) {
        v16h a  = load_A(&sh1[0][0], 256, kt * 32, lane);
        v16h bm = load_B(W2p, w * 8 + kt, lane);
        acc = __builtin_amdgcn_wmma_f32_16x16x32_f16(false, a, false, bm,
                                                     (short)0, acc, false, false);
      }
      int col = w * 16 + ncol;
#pragma unroll
      for (int r = 0; r < 8; ++r) {
        float v = acc[r] + sb2[col];
        semb[row0 + r][col] = (_Float16)(v > 0.0f ? v : 0.0f);
      }
    }
    __syncthreads();

    // ---- GEMM3: sg = semb@Wih + h@Whh + (bih+bhh)   N=256 -----------------
    {
      v8f acc = {0.f,0.f,0.f,0.f,0.f,0.f,0.f,0.f};
#pragma unroll
      for (int kt = 0; kt < 4; ++kt) {            // K=128 over emb
        v16h a  = load_A(&semb[0][0], 128, kt * 32, lane);
        v16h bm = load_B(Wip, w * 4 + kt, lane);
        acc = __builtin_amdgcn_wmma_f32_16x16x32_f16(false, a, false, bm,
                                                     (short)0, acc, false, false);
      }
#pragma unroll
      for (int kt = 0; kt < 2; ++kt) {            // K=64 recurrent over h
        v16h a  = load_A(&shh[0][0], 64, kt * 32, lane);
        v16h bm = load_B(Whp, w * 2 + kt, lane);
        acc = __builtin_amdgcn_wmma_f32_16x16x32_f16(false, a, false, bm,
                                                     (short)0, acc, false, false);
      }
      int col = w * 16 + ncol;
#pragma unroll
      for (int r = 0; r < 8; ++r) sg[row0 + r][col] = acc[r] + sbg[col];
    }
    __syncthreads();

    // ---- LSTM cell update (i,f,g,o split along the 256 gate columns) ------
    for (int s = tid; s < 16 * 64; s += 512) {
      int r = s >> 6, c = s & 63;
      float ig = sigmoidf_(sg[r][c]);
      float fg = sigmoidf_(sg[r][64 + c]);
      float gg = tanhf(sg[r][128 + c]);
      float og = sigmoidf_(sg[r][192 + c]);
      float cn = fg * sc[r][c] + ig * gg;
      sc[r][c]  = cn;
      shh[r][c] = (_Float16)(og * tanhf(cn));
    }
    __syncthreads();

    // ---- value head: wave w reduces row w (h . Wv + bv) -------------------
    {
      float s = (float)shh[w][lane] * sWv[lane] +
                (float)shh[w][lane + 32] * sWv[lane + 32];
#pragma unroll
      for (int off = 16; off > 0; off >>= 1) s += __shfl_xor(s, off, 32);
      if (lane == 0) out[(size_t)(b0 + w) * S_LEN + t] = s + sbv;
    }
    __syncthreads();
  }

  // ---- final hn / cn ------------------------------------------------------
  float* hn_out = out + (size_t)B_TOT * S_LEN;
  float* cn_out = hn_out + (size_t)B_TOT * HID;
  for (int s = tid; s < 16 * 64; s += 512) {
    int r = s >> 6, c = s & 63;
    hn_out[(size_t)(b0 + r) * HID + c] = (float)shh[r][c];
    cn_out[(size_t)(b0 + r) * HID + c] = sc[r][c];
  }
}

extern "C" void kernel_launch(void* const* d_in, const int* in_sizes, int n_in,
                              void* d_out, int out_size, void* d_ws, size_t ws_size,
                              hipStream_t stream) {
  const float* self_obs = (const float*)d_in[0];
  const float* tm_obs   = (const float*)d_in[1];
  const float* en_obs   = (const float*)d_in[2];
  const float* cp_obs   = (const float*)d_in[3];
  const float* W1  = (const float*)d_in[4];
  const float* b1  = (const float*)d_in[5];
  const float* W2  = (const float*)d_in[6];
  const float* b2  = (const float*)d_in[7];
  const float* Wih = (const float*)d_in[8];
  const float* bih = (const float*)d_in[9];
  const float* Whh = (const float*)d_in[10];
  const float* bhh = (const float*)d_in[11];
  const float* Wv  = (const float*)d_in[12];
  const float* bv  = (const float*)d_in[13];

  _Float16* wpk = (_Float16*)d_ws;
  float* out = (float*)d_out;

  // Pack weights to f16 WMMA-B fragment order (done once per call; tiny)
  pack_b_f16<<<(64 * 256 + 255) / 256, 256, 0, stream>>>(W1,  wpk + OFF_W1P,   47,  64, 256);
  pack_b_f16<<<(256 * 128 + 255) / 256, 256, 0, stream>>>(W2,  wpk + OFF_W2P,  256, 256, 128);
  pack_b_f16<<<(128 * 256 + 255) / 256, 256, 0, stream>>>(Wih, wpk + OFF_WIHP, 128, 128, 256);
  pack_b_f16<<<(64 * 256 + 255) / 256, 256, 0, stream>>>(Whh, wpk + OFF_WHHP,  64,  64, 256);

  // Raise dynamic-LDS cap for the 192 KB weight stage (CDNA5: 320 KB/WGP).
  // Ignore the result: if the runtime already allows it this is a no-op.
  (void)hipFuncSetAttribute((const void*)pod_critic_fused,
                            hipFuncAttributeMaxDynamicSharedMemorySize,
                            WPK_BYTES);

  pod_critic_fused<<<B_TOT / 16, 512, WPK_BYTES, stream>>>(
      self_obs, tm_obs, en_obs, cp_obs,
      b1, b2, bih, bhh, Wv, bv, wpk, out);
}
